// LSTMForecastModel_72670846649197
// MI455X (gfx1250) — matrix operand order repaired and checked by reading, not compiled, not run on hardware
//
#include <hip/hip_runtime.h>

// ---------------------------------------------------------------------------
// Fused seq2seq LSTM forecaster for MI455X (gfx1250, wave32, WMMA).
//   B=4096 T=128 OUT=64 H=200.  Each workgroup owns BM=32 batch rows and runs
//   the whole model (enc L0/L1 x128 steps, dec L0/L1 x64 steps + linear head)
//   with h/c state resident in LDS.  Recurrent GEMMs use
//   v_wmma_f32_16x16x32_bf16 with fp32 accumulation.  The wave's 7 N-tiles
//   are processed in blocks of {3,3,1} with K-outer/N-inner inside a block:
//   A-fragments (h, LDS) are loaded once per K-chunk per block and reused
//   across the block's N-tiles in registers; B-fragments (weights) stream
//   directly from L2 (all weights ~2.4MB bf16 << 192MB L2).  A laundered
//   zero OFFSET (not pointer) is added to the weight base each time step:
//   blocks LICM from hoisting the loop-invariant weight loads (which caused
//   scratch spills) while preserving global address-space provenance so the
//   loads lower to global_load_b128 (LOADcnt only), not flat (LOAD+DScnt).
// ---------------------------------------------------------------------------

#define DEVINL __device__ __forceinline__

constexpr int Bsz     = 4096;
constexpr int T       = 128;
constexpr int OUTLEN  = 64;
constexpr int H       = 200;
constexpr int G       = 4 * H;        // 800 real gate columns
constexpr int GP      = 896;          // gate columns padded to 56 N-tiles
constexpr int KP      = 224;          // H padded to 7*32 (WMMA K chunks)
constexpr int KC      = KP / 32;      // 7 K-chunks per GEMM pass
constexpr int NTHREADS = 256;         // 8 wave32
constexpr int NWAVES  = NTHREADS / 32;
constexpr int NPW     = (GP / 16) / NWAVES;  // 7 N-tiles per wave, uniform
constexpr int BM      = 32;           // batch rows per workgroup (2 M-tiles)

typedef __bf16 bf16x16 __attribute__((ext_vector_type(16)));
typedef __bf16 bf16x8  __attribute__((ext_vector_type(8)));
typedef float  f32x8   __attribute__((ext_vector_type(8)));

// LDS layout (dynamic):
//   gates : BM*GP  fp32   (114688 B)
//   c1,c2 : BM*H   fp32   ( 25600 B each)
//   h1,h2 : BM*KP  bf16   ( 14336 B each)  -- K-pad columns stay zero
//   staged consts: 6*G + KP + 2*BM fp32    ( 20352 B)
constexpr size_t SMEM_BYTES =
    (size_t)BM * GP * 4 + 2 * (size_t)BM * H * 4 + 2 * (size_t)BM * KP * 2 +
    (size_t)(6 * G + KP + 2 * BM) * 4;   // = 214912 B < 320KB WGP LDS

DEVINL float sigf(float v) { return 1.0f / (1.0f + __expf(-v)); }

// Compiler-opaque zero: adding it to a pointer blocks LICM/CSE of loads
// across the surrounding loop WITHOUT destroying address-space provenance
// (the pointer stays a GEP off the global kernarg -> global_load, not flat).
DEVINL size_t launder_zero() {
  size_t z = 0;
  asm volatile("" : "+s"(z));
  return z;
}

DEVINL f32x8 wmma_bf16(bf16x16 a, bf16x16 b, f32x8 c) {
  // D = A(16xK) * B(Kx16) + C ; fp32 accumulate
  return __builtin_amdgcn_wmma_f32_16x16x32_bf16(
      /*neg_a=*/false, a, /*neg_b=*/false, b,
      /*c_mod=*/(short)0, c, /*reuse_a=*/false, /*reuse_b=*/false);
}

// A-fragment: lane holds 16 contiguous K elements (k = 16*half + s).
DEVINL bf16x16 load_a(const __bf16* p) {
  bf16x8 lo = *(const bf16x8*)(p);
  bf16x8 hi = *(const bf16x8*)(p + 8);
  return __builtin_shufflevector(lo, hi, 0, 1, 2, 3, 4, 5, 6, 7,
                                 8, 9, 10, 11, 12, 13, 14, 15);
}

// One GEMM pass over a block of NBLK N-tiles (i0..i0+NBLK-1 of this wave):
//   acc[m][i] += A(m-tile) @ W(n-tile)^T, A register-resident per K-chunk.
template <int NBLK>
DEVINL void gemm_block(const __bf16* __restrict__ W,   // GP x KP bf16 (padded)
                       const __bf16* __restrict__ A,   // LDS BM x KP bf16
                       f32x8 (&acc0)[NBLK], f32x8 (&acc1)[NBLK],
                       int i0, int wave, int cl, int hi) {
  const __bf16* a0p = A + (size_t)cl * KP + hi * 16;
  const __bf16* a1p = A + (size_t)(16 + cl) * KP + hi * 16;
  // n = wave + (i0+i)*NWAVES  ->  weight row (n*16 + cl)
  const __bf16* wp =
      W + (size_t)((wave + i0 * NWAVES) * 16 + cl) * KP + hi * 16;
#pragma unroll
  for (int kc = 0; kc < KC; ++kc) {
    const bf16x16 a0 = load_a(a0p + kc * 32);
    const bf16x16 a1 = load_a(a1p + kc * 32);
#pragma unroll
    for (int i = 0; i < NBLK; ++i) {
      const bf16x16 b =
          *(const bf16x16*)(wp + (size_t)i * (NWAVES * 16) * KP + kc * 32);
      acc0[i] = wmma_bf16(a0, b, acc0[i]);
      acc1[i] = wmma_bf16(a1, b, acc1[i]);
    }
  }
}

// Compute + store gates for a block of NBLK N-tiles.
template <int NBLK, bool DUAL>
DEVINL void gate_block(const __bf16* __restrict__ W1,
                       const __bf16* __restrict__ W2,
                       const __bf16* __restrict__ A1,
                       const __bf16* __restrict__ A2,
                       float* __restrict__ gates,
                       int i0, int wave, int cl, int hi) {
  f32x8 acc0[NBLK] = {};
  f32x8 acc1[NBLK] = {};
  gemm_block<NBLK>(W1, A1, acc0, acc1, i0, wave, cl, hi);
  if (DUAL) gemm_block<NBLK>(W2, A2, acc0, acc1, i0, wave, cl, hi);

  // D layout: VGPR v, lanes0-15 -> M=v, lanes16-31 -> M=v+8 ; N = lane&15
#pragma unroll
  for (int i = 0; i < NBLK; ++i) {
    const int n = wave + (i0 + i) * NWAVES;
    float* g0 = gates + (size_t)(hi * 8) * GP + n * 16 + cl;
    float* g1 = gates + (size_t)(16 + hi * 8) * GP + n * 16 + cl;
#pragma unroll
    for (int v = 0; v < 8; ++v) {
      g0[(size_t)v * GP] = acc0[i][v];
      g1[(size_t)v * GP] = acc1[i][v];
    }
  }
}

// One LSTM cell step for the workgroup's BM x H state.
//   gates = A1 @ W1^T (+ A2 @ W2^T if DUAL)  via WMMA (bf16 in, fp32 acc)
//   then pointwise: c = sig(f)*c + sig(i)*tanh(g); h = sig(o)*tanh(c)
//   SCALAR adds x_row * Wih[:,0] (rank-1 input term) during activation.
template <bool SCALAR, bool DUAL>
DEVINL void lstm_step(const __bf16* W1,                // GP x KP bf16
                      const __bf16* W2,                // GP x KP bf16 or unused
                      const __bf16* __restrict__ A1,   // LDS BM x KP
                      const __bf16* __restrict__ A2,   // LDS BM x KP or unused
                      float* __restrict__ gates,       // LDS BM x GP
                      float* __restrict__ c,           // LDS BM x H
                      __bf16* __restrict__ h,          // LDS BM x KP (output)
                      const float* __restrict__ bias,  // LDS G
                      const float* __restrict__ wih,   // LDS G (SCALAR only)
                      const float* __restrict__ xin,   // LDS BM (SCALAR only)
                      int tid) {
  const int lane = tid & 31;
  const int wave = tid >> 5;
  const int cl = lane & 15;   // column within tile / row within A tile
  const int hi = lane >> 4;   // 0/1 half-lane group

  // Opaque zero offset per step: weight loads can't be hoisted across steps,
  // but provenance (global AS) is preserved -> global_load_b128.
  W1 += launder_zero();
  if (DUAL) W2 += launder_zero();

  // 7 N-tiles per wave in blocks of {3,3,1}: bounded register pressure.
  gate_block<3, DUAL>(W1, W2, A1, A2, gates, 0, wave, cl, hi);
  gate_block<3, DUAL>(W1, W2, A1, A2, gates, 3, wave, cl, hi);
  gate_block<1, DUAL>(W1, W2, A1, A2, gates, 6, wave, cl, hi);
  __syncthreads();

  // ---- gate activations + state update (fp32) ----
  for (int u = tid; u < BM * H; u += NTHREADS) {
    const int r = u / H;
    const int j = u - r * H;
    const float* gr = gates + (size_t)r * GP;
    float ai = bias[j], af = bias[H + j], ag = bias[2 * H + j],
          ao = bias[3 * H + j];
    if (SCALAR) {
      const float xv = xin[r];
      ai += xv * wih[j];
      af += xv * wih[H + j];
      ag += xv * wih[2 * H + j];
      ao += xv * wih[3 * H + j];
    }
    const float gi = gr[j] + ai;
    const float gf = gr[H + j] + af;
    const float gg = gr[2 * H + j] + ag;
    const float go = gr[3 * H + j] + ao;
    const float cn = sigf(gf) * c[u] + sigf(gi) * tanhf(gg);
    c[u] = cn;
    h[(size_t)r * KP + j] = (__bf16)(sigf(go) * tanhf(cn));
  }
  __syncthreads();
}

__global__ void __launch_bounds__(NTHREADS)
lstm_forecast_kernel(const float* __restrict__ x,
                     const float* __restrict__ enc_Wih0,
                     const float* __restrict__ enc_b0,
                     const float* __restrict__ enc_b1,
                     const float* __restrict__ dec_Wih0,
                     const float* __restrict__ dec_b0,
                     const float* __restrict__ dec_b1,
                     const float* __restrict__ lin_W,
                     const float* __restrict__ lin_b,
                     const __bf16* __restrict__ Wpack,  // 6x (GP*KP) bf16
                     float* __restrict__ out) {
  extern __shared__ char smem[];
  float* gates = (float*)smem;                       // BM*GP
  float* c1 = gates + (size_t)BM * GP;               // BM*H
  float* c2 = c1 + (size_t)BM * H;                   // BM*H
  __bf16* h1 = (__bf16*)(c2 + (size_t)BM * H);       // BM*KP
  __bf16* h2 = h1 + (size_t)BM * KP;                 // BM*KP
  float* sb = (float*)(h2 + (size_t)BM * KP);
  float* s_eb0 = sb;
  float* s_ewih0 = sb + G;
  float* s_eb1 = sb + 2 * G;
  float* s_db0 = sb + 3 * G;
  float* s_dwih0 = sb + 4 * G;
  float* s_db1 = sb + 5 * G;
  float* s_linw = sb + 6 * G;          // KP floats (pad unused)
  float* s_pred = s_linw + KP;         // BM
  float* s_xcur = s_pred + BM;         // BM

  const int tid = threadIdx.x;
  const int b0 = blockIdx.x * BM;

  // init state + stage small constants
  for (int i = tid; i < BM * KP; i += NTHREADS) {
    h1[i] = (__bf16)0.0f;
    h2[i] = (__bf16)0.0f;
  }
  for (int i = tid; i < BM * H; i += NTHREADS) {
    c1[i] = 0.0f;
    c2[i] = 0.0f;
  }
  for (int i = tid; i < G; i += NTHREADS) {
    s_eb0[i] = enc_b0[i];
    s_ewih0[i] = enc_Wih0[i];
    s_eb1[i] = enc_b1[i];
    s_db0[i] = dec_b0[i];
    s_dwih0[i] = dec_Wih0[i];
    s_db1[i] = dec_b1[i];
  }
  for (int i = tid; i < H; i += NTHREADS) s_linw[i] = lin_W[i];
  const float linb = lin_b[0];
  __syncthreads();

  const size_t MSZ = (size_t)GP * KP;
  const __bf16* W_eWhh0 = Wpack + 0 * MSZ;
  const __bf16* W_eWih1 = Wpack + 1 * MSZ;
  const __bf16* W_eWhh1 = Wpack + 2 * MSZ;
  const __bf16* W_dWhh0 = Wpack + 3 * MSZ;
  const __bf16* W_dWih1 = Wpack + 4 * MSZ;
  const __bf16* W_dWhh1 = Wpack + 5 * MSZ;

  // ---------------- encoder: 128 steps, two fused layers ----------------
  for (int t = 0; t < T; ++t) {
    if (tid < BM) s_xcur[tid] = x[(size_t)(b0 + tid) * T + t];
    // layer0: gates = h1@Whh0^T + x_t*Wih0 + b0
    lstm_step<true, false>(W_eWhh0, nullptr, h1, nullptr, gates, c1, h1,
                           s_eb0, s_ewih0, s_xcur, tid);
    // layer1: gates = h1@Wih1^T + h2@Whh1^T + b1
    lstm_step<false, true>(W_eWih1, W_eWhh1, h1, h2, gates, c2, h2,
                           s_eb1, nullptr, nullptr, tid);
  }

  // ---------------- decoder: 64 steps with linear feedback ----------------
  if (tid < BM) s_pred[tid] = x[(size_t)(b0 + tid) * T + (T - 1)];
  __syncthreads();
  for (int s = 0; s < OUTLEN; ++s) {
    lstm_step<true, false>(W_dWhh0, nullptr, h1, nullptr, gates, c1, h1,
                           s_db0, s_dwih0, s_pred, tid);
    lstm_step<false, true>(W_dWih1, W_dWhh1, h1, h2, gates, c2, h2,
                           s_db1, nullptr, nullptr, tid);
    if (tid < BM) {
      float acc = linb;
      const __bf16* hr = h2 + (size_t)tid * KP;
      for (int j = 0; j < H; ++j) acc += (float)hr[j] * s_linw[j];
      s_pred[tid] = acc;
      out[(size_t)(b0 + tid) * OUTLEN + s] = acc;  // preds.T -> (B, OUT)
    }
    __syncthreads();
  }
}

// Pack the 6 recurrent weight matrices (G x H fp32, row-major) into
// N- and K-padded bf16 (GP x KP): K-pad multiplies zeroed h columns,
// N-pad rows (800..895) are zero so every wave owns exactly 7 N-tiles.
__global__ void pack_weights(const float* __restrict__ s0,
                             const float* __restrict__ s1,
                             const float* __restrict__ s2,
                             const float* __restrict__ s3,
                             const float* __restrict__ s4,
                             const float* __restrict__ s5,
                             __bf16* __restrict__ dst) {
  const int total = 6 * GP * KP;
  int i = blockIdx.x * blockDim.x + threadIdx.x;
  if (i >= total) return;
  const int mat = i / (GP * KP);
  const int rem = i - mat * (GP * KP);
  const int n = rem / KP;
  const int k = rem - n * KP;
  const float* srcs[6] = {s0, s1, s2, s3, s4, s5};
  const float v = (n < G && k < H) ? srcs[mat][n * H + k] : 0.0f;
  dst[i] = (__bf16)v;
}

extern "C" void kernel_launch(void* const* d_in, const int* in_sizes, int n_in,
                              void* d_out, int out_size, void* d_ws,
                              size_t ws_size, hipStream_t stream) {
  (void)in_sizes; (void)n_in; (void)out_size; (void)ws_size;
  const float* x        = (const float*)d_in[0];
  const float* enc_Wih0 = (const float*)d_in[1];
  const float* enc_Whh0 = (const float*)d_in[2];
  const float* enc_b0   = (const float*)d_in[3];
  const float* enc_Wih1 = (const float*)d_in[4];
  const float* enc_Whh1 = (const float*)d_in[5];
  const float* enc_b1   = (const float*)d_in[6];
  const float* dec_Wih0 = (const float*)d_in[7];
  const float* dec_Whh0 = (const float*)d_in[8];
  const float* dec_b0   = (const float*)d_in[9];
  const float* dec_Wih1 = (const float*)d_in[10];
  const float* dec_Whh1 = (const float*)d_in[11];
  const float* dec_b1   = (const float*)d_in[12];
  const float* lin_W    = (const float*)d_in[13];
  const float* lin_b    = (const float*)d_in[14];
  __bf16* Wpack = (__bf16*)d_ws;  // 6 * 896 * 224 bf16 = 2.41 MB

  const int total = 6 * GP * KP;
  pack_weights<<<(total + 255) / 256, 256, 0, stream>>>(
      enc_Whh0, enc_Wih1, enc_Whh1, dec_Whh0, dec_Wih1, dec_Whh1, Wpack);

  hipFuncSetAttribute((const void*)lstm_forecast_kernel,
                      hipFuncAttributeMaxDynamicSharedMemorySize,
                      (int)SMEM_BYTES);
  lstm_forecast_kernel<<<Bsz / BM, NTHREADS, SMEM_BYTES, stream>>>(
      x, enc_Wih0, enc_b0, enc_b1, dec_Wih0, dec_b0, dec_b1, lin_W, lin_b,
      Wpack, (float*)d_out);
}
